// HRNNV2Est3C1_50740743635506
// MI455X (gfx1250) — matrix operand order repaired and checked by reading, not compiled, not run on hardware
//
#include <hip/hip_runtime.h>
#include <math.h>

// ---------------- types ----------------
typedef __bf16 bf16_t;
typedef bf16_t v8bf  __attribute__((ext_vector_type(8)));
typedef bf16_t v16bf __attribute__((ext_vector_type(16)));
typedef float  v8f   __attribute__((ext_vector_type(8)));

union FragA { v16bf v; v8bf h[2]; };

__device__ __forceinline__ v8f wmma_bf16(v16bf a, v16bf b, v8f c) {
    return __builtin_amdgcn_wmma_f32_16x16x32_bf16(false, a, false, b, (short)0, c, false, false);
}
__device__ __forceinline__ float sigm(float x) { return 1.0f / (1.0f + __expf(-x)); }

// problem constants
#define BSZ 512
#define TSTEPS 512
#define FDIM 144
#define HDIM 512
#define SR 1184      // padded rcat width: [x(144) pad(16) h_a(512) h_b(512)]
#define KA 672       // padded cellA K: [x(144) pad(16) h_a(512)]

// ---------------- pre-pass kernels ----------------
__global__ void k_conv(bf16_t* __restrict__ dst, const float* __restrict__ src, int n) {
    int i = blockIdx.x * 256 + threadIdx.x;
    if (i < n) dst[i] = (bf16_t)src[i];
}

// W1 (1024 x 1168) -> w1p (1024 x 1184) with 16 zero cols inserted at col 144
__global__ void k_w1p(bf16_t* __restrict__ dst, const float* __restrict__ W1) {
    int i = blockIdx.x * 256 + threadIdx.x;
    if (i >= 1024 * SR) return;
    int r = i / SR, k = i - r * SR;
    float v;
    if (k < 144)      v = W1[r * 1168 + k];
    else if (k < 160) v = 0.0f;
    else              v = W1[r * 1168 + (k - 16)];
    dst[i] = (bf16_t)v;
}

// [Wih_a (2048x144) | zeros(16) | Whh_a (2048x512)] -> wacat (2048 x 672)
__global__ void k_wacat(bf16_t* __restrict__ dst, const float* __restrict__ Wih,
                        const float* __restrict__ Whh) {
    int i = blockIdx.x * 256 + threadIdx.x;
    if (i >= 2048 * KA) return;
    int r = i / KA, k = i - r * KA;
    float v;
    if (k < 144)      v = Wih[r * 144 + k];
    else if (k < 160) v = 0.0f;
    else              v = Whh[r * 512 + (k - 160)];
    dst[i] = (bf16_t)v;
}

// x0 (B,C,P,T)=(b, f, t) strided -> xT[t][b][f] bf16 (coalesced per-step loads)
__global__ void k_xT(bf16_t* __restrict__ dst, const float* __restrict__ x0) {
    int i = blockIdx.x * 256 + threadIdx.x;   // i = b*F + f
    if (i >= BSZ * FDIM) return;
    int b = i / FDIM, f = i - b * FDIM;
    const float* src = x0 + (size_t)i * TSTEPS;
    for (int t = 0; t < TSTEPS; ++t)
        dst[((size_t)t * BSZ + b) * FDIM + f] = (bf16_t)src[t];
}

// ---------------- main persistent-scan kernel ----------------
// grid = 32 blocks (16 batch rows each), block = 512 threads = 16 waves (wave32)
__global__ __launch_bounds__(512, 1)
void hrnn_main(const bf16_t* __restrict__ xT,
               const bf16_t* __restrict__ w1p,
               const bf16_t* __restrict__ wacat,
               const bf16_t* __restrict__ wihb,
               const bf16_t* __restrict__ whhb,
               const bf16_t* __restrict__ w2p,
               const float* __restrict__ biha, const float* __restrict__ bhha,
               const float* __restrict__ bihb, const float* __restrict__ bhhb,
               const float* __restrict__ b1,   const float* __restrict__ b2,
               const float* __restrict__ W3,   const float* __restrict__ b3,
               const float* __restrict__ Wc,   const float* __restrict__ bc,
               float* __restrict__ hbn_ws,
               float* __restrict__ out)
{
    __shared__ __attribute__((aligned(32))) bf16_t s_rcat[16 * SR];      // 37.9 KB
    __shared__ __attribute__((aligned(32))) bf16_t s_han[16 * HDIM];     // 16 KB
    __shared__ __attribute__((aligned(32))) bf16_t s_r1[16 * 1024];      // 32 KB (also temp h_a2/h_b2)
    __shared__ __attribute__((aligned(32))) bf16_t s_r2[16 * 256];       // 8 KB
    __shared__ int s_alpha[16];

    const int tid  = threadIdx.x;
    const int wave = tid >> 5;
    const int lane = tid & 31;
    const int ln   = lane & 15;     // A-row / B-col / C-col index
    const int lh   = lane >> 4;     // half-wave selector
    const int blk  = blockIdx.x;

    // init: rcat = [x_0 | 0...], c-state = 0
    for (int i = tid; i < 16 * SR; i += 512) s_rcat[i] = (bf16_t)0.0f;
    __syncthreads();
    for (int i = tid; i < 16 * FDIM; i += 512) {
        int m = i / FDIM, f = i - m * FDIM;
        s_rcat[m * SR + f] = xT[((size_t)0 * BSZ + blk * 16 + m) * FDIM + f];
    }
    __syncthreads();

    v8f ca[2] = {};   // persistent c_a (16 rows x 32 cols owned by this wave)
    v8f cb[2] = {};   // persistent c_b

    for (int t = 0; t < TSTEPS; ++t) {
        // ======= MLP layer 1: r1 = relu(rcat @ W1p^T + b1), 64 N-tiles / 16 waves =======
        for (int i = 0; i < 4; ++i) {
            int nt = wave * 4 + i;
            v8f acc = {};
            const bf16_t* wrow = w1p + (size_t)(nt * 16 + ln) * SR;
            for (int kc = 0; kc < 37; ++kc) {
                int k0 = kc * 32;
                FragA fa;
                fa.h[0] = *(const v8bf*)(s_rcat + ln * SR + k0 + lh * 16);
                fa.h[1] = *(const v8bf*)(s_rcat + ln * SR + k0 + lh * 16 + 8);
                v16bf fb = *(const v16bf*)(wrow + k0 + lh * 16);
                acc = wmma_bf16(fa.v, fb, acc);
            }
            float bias = b1[nt * 16 + ln];
            #pragma unroll
            for (int j = 0; j < 8; ++j) {
                int m = j + lh * 8;
                float v = acc[j] + bias;
                s_r1[m * 1024 + nt * 16 + ln] = (bf16_t)(v > 0.0f ? v : 0.0f);
            }
        }
        __syncthreads();

        // ======= MLP layer 2: r2 = relu(r1 @ W2^T + b2), 16 tiles / 16 waves =======
        {
            v8f acc = {};
            const bf16_t* wrow = w2p + (size_t)(wave * 16 + ln) * 1024;
            for (int kc = 0; kc < 32; ++kc) {
                int k0 = kc * 32;
                FragA fa;
                fa.h[0] = *(const v8bf*)(s_r1 + ln * 1024 + k0 + lh * 16);
                fa.h[1] = *(const v8bf*)(s_r1 + ln * 1024 + k0 + lh * 16 + 8);
                v16bf fb = *(const v16bf*)(wrow + k0 + lh * 16);
                acc = wmma_bf16(fa.v, fb, acc);
            }
            float bias = b2[wave * 16 + ln];
            #pragma unroll
            for (int j = 0; j < 8; ++j) {
                int m = j + lh * 8;
                float v = acc[j] + bias;
                s_r2[m * 256 + wave * 16 + ln] = (bf16_t)(v > 0.0f ? v : 0.0f);
            }
        }
        __syncthreads();

        // ======= logits / argmax / alpha (wave 0, one lane per batch row) =======
        if (wave == 0 && lh == 0) {
            float l0 = b3[0], l1 = b3[1], l2 = b3[2];
            const bf16_t* rr = s_r2 + ln * 256;
            for (int k = 0; k < 256; ++k) {
                float rv = (float)rr[k];
                l0 += rv * W3[k];
                l1 += rv * W3[256 + k];
                l2 += rv * W3[512 + k];
            }
            int idx = 0; float best = l0;
            if (l1 > best) { best = l1; idx = 1; }
            if (l2 > best) { best = l2; idx = 2; }
            s_alpha[ln] = idx;
            out[5120 + (size_t)(blk * 16 + ln) * TSTEPS + t] = (idx == 1) ? 1.0f : 0.0f;
        }
        __syncthreads();

        // ======= LSTM cell A: gates over [x_t|pad|h_a] @ [Wih_a|0|Whh_a]^T =======
        for (int sub = 0; sub < 2; ++sub) {
            int col = wave * 32 + sub * 16;
            v8f acc4[4] = {};
            for (int kc = 0; kc < 21; ++kc) {
                int k0 = kc * 32;
                FragA fa;
                fa.h[0] = *(const v8bf*)(s_rcat + ln * SR + k0 + lh * 16);
                fa.h[1] = *(const v8bf*)(s_rcat + ln * SR + k0 + lh * 16 + 8);
                #pragma unroll
                for (int g = 0; g < 4; ++g) {
                    v16bf fb = *(const v16bf*)(wacat + (size_t)(g * HDIM + col + ln) * KA + k0 + lh * 16);
                    acc4[g] = wmma_bf16(fa.v, fb, acc4[g]);
                }
            }
            int colf = col + ln;
            float bi  = biha[colf]            + bhha[colf];
            float bff = biha[HDIM + colf]     + bhha[HDIM + colf];
            float bg  = biha[2 * HDIM + colf] + bhha[2 * HDIM + colf];
            float bo  = biha[3 * HDIM + colf] + bhha[3 * HDIM + colf];
            #pragma unroll
            for (int j = 0; j < 8; ++j) {
                int m = j + lh * 8;
                float iv = sigm(acc4[0][j] + bi);
                float fv = sigm(acc4[1][j] + bff);
                float gv = tanhf(acc4[2][j] + bg);
                float ov = sigm(acc4[3][j] + bo);
                float cold = ca[sub][j];
                float cn = fv * cold + iv * gv;
                float hn = ov * tanhf(cn);
                int aidx = s_alpha[m];
                float a1f = (aidx == 1) ? 1.0f : 0.0f;
                float a2f = (aidx == 2) ? 1.0f : 0.0f;
                ca[sub][j] = a1f * cn + a2f * cold;
                float hold = (float)s_rcat[m * SR + 160 + colf];
                s_han[m * HDIM + colf] = (bf16_t)hn;                          // h_a_n (cell B input)
                s_r1[m * HDIM + colf]  = (bf16_t)(a1f * hn + a2f * hold);     // temp h_a2
            }
        }
        __syncthreads();

        // ======= LSTM cell B: gates = h_a_n @ Wih_b^T + h_b @ Whh_b^T =======
        for (int sub = 0; sub < 2; ++sub) {
            int col = wave * 32 + sub * 16;
            v8f acc4[4] = {};
            for (int kc = 0; kc < 16; ++kc) {             // K over h_a_n
                int k0 = kc * 32;
                FragA fa;
                fa.h[0] = *(const v8bf*)(s_han + ln * HDIM + k0 + lh * 16);
                fa.h[1] = *(const v8bf*)(s_han + ln * HDIM + k0 + lh * 16 + 8);
                #pragma unroll
                for (int g = 0; g < 4; ++g) {
                    v16bf fb = *(const v16bf*)(wihb + (size_t)(g * HDIM + col + ln) * HDIM + k0 + lh * 16);
                    acc4[g] = wmma_bf16(fa.v, fb, acc4[g]);
                }
            }
            for (int kc = 0; kc < 16; ++kc) {             // K over h_b (old)
                int k0 = kc * 32;
                FragA fa;
                fa.h[0] = *(const v8bf*)(s_rcat + ln * SR + 672 + k0 + lh * 16);
                fa.h[1] = *(const v8bf*)(s_rcat + ln * SR + 672 + k0 + lh * 16 + 8);
                #pragma unroll
                for (int g = 0; g < 4; ++g) {
                    v16bf fb = *(const v16bf*)(whhb + (size_t)(g * HDIM + col + ln) * HDIM + k0 + lh * 16);
                    acc4[g] = wmma_bf16(fa.v, fb, acc4[g]);
                }
            }
            int colf = col + ln;
            float bi  = bihb[colf]            + bhhb[colf];
            float bff = bihb[HDIM + colf]     + bhhb[HDIM + colf];
            float bg  = bihb[2 * HDIM + colf] + bhhb[2 * HDIM + colf];
            float bo  = bihb[3 * HDIM + colf] + bhhb[3 * HDIM + colf];
            #pragma unroll
            for (int j = 0; j < 8; ++j) {
                int m = j + lh * 8;
                float iv = sigm(acc4[0][j] + bi);
                float fv = sigm(acc4[1][j] + bff);
                float gv = tanhf(acc4[2][j] + bg);
                float ov = sigm(acc4[3][j] + bo);
                float cold = cb[sub][j];
                float cn = fv * cold + iv * gv;
                float hn = ov * tanhf(cn);
                int aidx = s_alpha[m];
                float a0f = (aidx == 0) ? 1.0f : 0.0f;
                cb[sub][j] = a0f * cn + (1.0f - a0f) * cold;
                float hold = (float)s_rcat[m * SR + 672 + colf];
                s_r1[8192 + m * HDIM + colf] = (bf16_t)(a0f * hn + (1.0f - a0f) * hold); // temp h_b2
                if (t == TSTEPS - 1)
                    hbn_ws[(size_t)(blk * 16 + m) * HDIM + colf] = hn;  // last_hb = h_b_n (ungated)
            }
        }
        __syncthreads();

        // ======= assemble rcat for step t+1 =======
        for (int i = tid; i < 16 * HDIM; i += 512) {
            int m = i >> 9, k = i & 511;
            s_rcat[m * SR + 160 + k] = s_r1[i];
            s_rcat[m * SR + 672 + k] = s_r1[8192 + i];
        }
        if (t + 1 < TSTEPS) {
            for (int i = tid; i < 16 * FDIM; i += 512) {
                int m = i / FDIM, f = i - m * FDIM;
                s_rcat[m * SR + f] = xT[((size_t)(t + 1) * BSZ + blk * 16 + m) * FDIM + f];
            }
        }
        __syncthreads();
    }

    // ======= x_out = last_hb @ Wc^T + bc  (16 rows x 10 cols per block) =======
    __threadfence();
    __syncthreads();
    if (tid < 160) {
        int m = tid / 10, c = tid - m * 10;
        float acc = bc[c];
        const float* hp = hbn_ws + (size_t)(blk * 16 + m) * HDIM;
        const float* wp = Wc + c * HDIM;
        for (int k = 0; k < HDIM; ++k) acc += hp[k] * wp[k];
        out[(size_t)(blk * 16 + m) * 10 + c] = acc;
    }
}

// ---------------- launcher ----------------
extern "C" void kernel_launch(void* const* d_in, const int* in_sizes, int n_in,
                              void* d_out, int out_size, void* d_ws, size_t ws_size,
                              hipStream_t stream) {
    const float* x0    = (const float*)d_in[0];
    const float* Wih_a = (const float*)d_in[1];
    const float* Whh_a = (const float*)d_in[2];
    const float* bih_a = (const float*)d_in[3];
    const float* bhh_a = (const float*)d_in[4];
    const float* Wih_b = (const float*)d_in[5];
    const float* Whh_b = (const float*)d_in[6];
    const float* bih_b = (const float*)d_in[7];
    const float* bhh_b = (const float*)d_in[8];
    const float* W1    = (const float*)d_in[9];
    const float* b1    = (const float*)d_in[10];
    const float* W2    = (const float*)d_in[11];
    const float* b2    = (const float*)d_in[12];
    const float* W3    = (const float*)d_in[13];
    const float* b3    = (const float*)d_in[14];
    const float* Wc    = (const float*)d_in[15];
    const float* bc    = (const float*)d_in[16];
    float* out = (float*)d_out;

    char* ws = (char*)d_ws;
    size_t off = 0;
    auto carve = [&](size_t bytes) { size_t r = off; off += (bytes + 255) & ~(size_t)255; return r; };
    bf16_t* w1p   = (bf16_t*)(ws + carve((size_t)1024 * SR * 2));
    bf16_t* wacat = (bf16_t*)(ws + carve((size_t)2048 * KA * 2));
    bf16_t* wihb  = (bf16_t*)(ws + carve((size_t)2048 * 512 * 2));
    bf16_t* whhb  = (bf16_t*)(ws + carve((size_t)2048 * 512 * 2));
    bf16_t* w2p   = (bf16_t*)(ws + carve((size_t)256 * 1024 * 2));
    bf16_t* xT    = (bf16_t*)(ws + carve((size_t)TSTEPS * BSZ * FDIM * 2));
    float*  hbn   = (float*) (ws + carve((size_t)BSZ * HDIM * 4));
    (void)ws_size; (void)in_sizes; (void)n_in; (void)out_size;

    k_w1p  <<<(1024 * SR + 255) / 256, 256, 0, stream>>>(w1p, W1);
    k_wacat<<<(2048 * KA + 255) / 256, 256, 0, stream>>>(wacat, Wih_a, Whh_a);
    k_conv <<<(2048 * 512 + 255) / 256, 256, 0, stream>>>(wihb, Wih_b, 2048 * 512);
    k_conv <<<(2048 * 512 + 255) / 256, 256, 0, stream>>>(whhb, Whh_b, 2048 * 512);
    k_conv <<<(256 * 1024 + 255) / 256, 256, 0, stream>>>(w2p, W2, 256 * 1024);
    k_xT   <<<(BSZ * FDIM + 255) / 256, 256, 0, stream>>>(xT, x0);

    hrnn_main<<<BSZ / 16, 512, 0, stream>>>(xT, w1p, wacat, wihb, whhb, w2p,
                                            bih_a, bhh_a, bih_b, bhh_b,
                                            b1, b2, W3, b3, Wc, bc,
                                            hbn, out);
}
